// DVAE_RBM_7902739824804
// MI455X (gfx1250) — matrix operand order repaired and checked by reading, not compiled, not run on hardware
//
#include <hip/hip_runtime.h>
#include <cstdint>
#include <cstddef>

typedef unsigned short u16;
typedef __attribute__((ext_vector_type(16))) __bf16 v16bf;
typedef __attribute__((ext_vector_type(16))) unsigned short v16u;
typedef __attribute__((ext_vector_type(8)))  float v8f;
typedef __attribute__((ext_vector_type(4)))  int v4i;

#define BSZ   16384
#define DIN   4000
#define HDIM  512
#define LDIM  256
#define NSTEP 30
#define EXPB1 0.6487212707001282f   // exp(0.5)-1
#define INV_BETA 2.0f               // 1/0.5

#define BM  128
#define BN  64
#define NT  (BN/16)
#define LDA 40        // padded LDS row stride (u16 elems): 80B, 16B-aligned, bank-staggered
#define GLDW 264      // gibbs LDS row stride (u16): 528B, 16B-aligned, bank-staggered

#if defined(__HIP_DEVICE_COMPILE__) && __has_builtin(__builtin_amdgcn_global_load_async_to_lds_b128)
#define USE_ASYNC 1
#else
#define USE_ASYNC 0
#endif

#define AS1 __attribute__((address_space(1)))
#define AS3 __attribute__((address_space(3)))

union Frag { v16u u; v16bf b; uint4 q[2]; };

__device__ inline u16 f2bf(float f){
  unsigned u = __float_as_uint(f);
  u += 0x7FFFu + ((u >> 16) & 1u);        // round-to-nearest-even
  return (u16)(u >> 16);
}
__device__ inline float bf2f(u16 b){ return __uint_as_float(((unsigned)b) << 16); }

__device__ inline v8f zero8(){ v8f z = {0.f,0.f,0.f,0.f,0.f,0.f,0.f,0.f}; return z; }

// 16B global -> LDS copy; async (ASYNCcnt path) when available.
__device__ inline void cp16(const u16* __restrict__ g, u16* __restrict__ l){
#if USE_ASYNC
  __builtin_amdgcn_global_load_async_to_lds_b128((AS1 v4i*)g, (AS3 v4i*)l, 0, 0);
#else
  *(uint4*)l = *(const uint4*)g;
#endif
}
__device__ inline void wait_async(){
#if USE_ASYNC
#if __has_builtin(__builtin_amdgcn_s_wait_asynccnt)
  __builtin_amdgcn_s_wait_asynccnt(0);
#else
  asm volatile("s_wait_asynccnt 0x0" ::: "memory");
#endif
#endif
}

__device__ inline v8f wmma_bf16(v16bf a, v16bf b, v8f c){
  return __builtin_amdgcn_wmma_f32_16x16x32_bf16(false, a, false, b, (short)0, c, false, false);
}

// A fragment: 16x32 bf16 tile stored row-major in LDS (stride LDA).
// lane<16 holds row=lane, K {0..7,16..23}; lane>=16 holds row=lane-16, K {8..15,24..31}.
__device__ inline v16bf load_a_frag(const u16* rowPtr, int lane){
  Frag f;
  const int off = (lane & 16) ? 8 : 0;
  f.q[0] = *(const uint4*)(rowPtr + off);
  f.q[1] = *(const uint4*)(rowPtr + 16 + off);
  return f.b;
}
// B fragment: 32x16 bf16 tile stored N-major in LDS ([n][k], stride LDA).
// lane<16 holds col=lane, K {0..7,16..23}; lane>=16 holds col=lane-16, K {8..15,24..31}.
__device__ inline v16bf load_b_frag(const u16* rowPtr, int lane){
  Frag f;
  const int off = (lane & 16) ? 8 : 0;
  f.q[0] = *(const uint4*)(rowPtr + off);
  f.q[1] = *(const uint4*)(rowPtr + 16 + off);
  return f.b;
}

struct EpiParams {
  float* out_f32;                 // MODE 0
  float* col_sum; float* col_sumsq;
  const float* rho;               // MODE 1
  float* z_out; float* zeta_out; u16* zeta_bf; u16* z_bf; float* logq_accum;
  const float* x_ref; float* recon_accum;            // MODE 2
  const u16* z_src; const float* rbm_h; float* energy_accum; // MODE 3
};

// W is pre-converted bf16, transposed to [N][K] row-major (see convert_transpose_kernel).
// MODE 0: store f32 + column stats (BN pre-pass)
// MODE 1: spike-and-exp reparameterization epilogue
// MODE 2: fused recon-loss reduction (no store of x_recon)
// MODE 3: RBM energy reduction sum z*(h + zW)
template<bool A_BF16, int MODE>
__global__ __launch_bounds__(256) void gemm_kernel(const void* __restrict__ Aglob,
    const u16* __restrict__ Wt, const float* __restrict__ bias,
    int M, int N, int K, EpiParams p)
{
  __shared__ u16 sA[BM * LDA];
  __shared__ u16 sB[BN * LDA];
  __shared__ float sSum[BN];
  __shared__ float sSq[BN];
  __shared__ float sRed;

  const int tid  = threadIdx.x;
  const int wave = tid >> 5;
  const int lane = tid & 31;
  const int colBlk = blockIdx.x * BN;   // fastest-varying -> A tile stays hot in L2
  const int rowBlk = blockIdx.y * BM;

  if (tid < BN){ sSum[tid] = 0.f; sSq[tid] = 0.f; }
  if (tid == 0) sRed = 0.f;

  v8f acc[NT];
  #pragma unroll
  for (int t = 0; t < NT; t++) acc[t] = zero8();

  const int r  = tid >> 1;      // A-tile row owned by this thread (BM=128, 2 thr/row)
  const int hh = tid & 1;       // half of the 32-wide K slab
  const int bn = tid >> 2;      // B-tile row (n) owned by this thread
  const int bc = tid & 3;       // 8-element K chunk

  for (int k0 = 0; k0 < K; k0 += 32){
    { // stage A tile (BM x 32) as bf16
      u16* dst = &sA[r * LDA + hh * 16];
      if (A_BF16){
        const u16* Ag = (const u16*)Aglob + (size_t)(rowBlk + r) * K + k0 + hh * 16;
        cp16(Ag, dst);
        cp16(Ag + 8, dst + 8);
        if (k0 + 32 < K) __builtin_prefetch(Ag + 32, 0, 0);
      } else {
        const float* Ag = (const float*)Aglob + (size_t)(rowBlk + r) * K + k0 + hh * 16;
        #pragma unroll
        for (int q2 = 0; q2 < 4; q2++){
          float4 f = *(const float4*)(Ag + q2 * 4);
          dst[q2*4+0] = f2bf(f.x); dst[q2*4+1] = f2bf(f.y);
          dst[q2*4+2] = f2bf(f.z); dst[q2*4+3] = f2bf(f.w);
        }
        if (k0 + 32 < K) __builtin_prefetch(Ag + 32, 0, 0);
      }
    }
    { // stage B tile: 64 n-rows x 32 k, straight 16B copies from [N][K] bf16 weights
      int gcol = colBlk + bn;
      u16* dstB = &sB[bn * LDA + bc * 8];
      if (gcol < N){
        const u16* srcB = Wt + (size_t)gcol * K + k0 + bc * 8;
        cp16(srcB, dstB);
      } else {
        uint4 zz = {0u,0u,0u,0u};
        *(uint4*)dstB = zz;
      }
    }
    wait_async();
    __syncthreads();
    {
      const u16* aPtr = &sA[((wave << 4) + (lane & 15)) * LDA];
      v16bf a = load_a_frag(aPtr, lane);
      #pragma unroll
      for (int t = 0; t < NT; t++){
        const u16* bPtr = &sB[((t << 4) + (lane & 15)) * LDA];
        acc[t] = wmma_bf16(a, load_b_frag(bPtr, lane), acc[t]);
      }
    }
    __syncthreads();
  }

  // ---- epilogue ----
  // C layout: lane 0-15 -> N=lane, rows M = vgpr i; lane 16-31 -> N=lane-16, rows M = i+8
  const int rbase = rowBlk + (wave << 4) + ((lane & 16) ? 8 : 0);
  const int nlo = lane & 15;
  float localAcc = 0.f;

  #pragma unroll
  for (int t = 0; t < NT; t++){
    int n = (t << 4) + nlo;
    int col = colBlk + n;
    if (col < N){
      float bv = bias ? bias[col] : 0.f;
      float ls = 0.f, lsq = 0.f;
      #pragma unroll
      for (int i = 0; i < 8; i++){
        int row = rbase + i;
        float v = acc[t][i] + bv;
        if constexpr (MODE == 0){
          p.out_f32[(size_t)row * N + col] = v;
          ls += v; lsq += v * v;
        } else if constexpr (MODE == 1){
          size_t idx = (size_t)row * N + col;
          float e   = __expf(-v);            // q = 1/(1+e), 1/q = 1+e
          float ope = 1.f + e;
          float q   = __builtin_amdgcn_rcpf(ope);
          float rr  = p.rho[idx];
          float omq = 1.f - q;
          float zeta = 0.f, z = 0.f;
          if (rr > omq){
            float arg = (rr - omq) * ope * EXPB1 + 1.f;
            zeta = __logf(arg) * INV_BETA;
            z = (zeta > 0.f) ? 1.f : 0.f;
          }
          p.z_out[idx]    = z;
          p.zeta_out[idx] = zeta;
          p.zeta_bf[idx]  = f2bf(zeta);
          p.z_bf[idx]     = (z > 0.f) ? (u16)0x3F80 : (u16)0;
          float qc = fminf(fmaxf(q, 1e-7f), 1.f - 1e-7f);
          localAcc += (z > 0.f) ? __logf(qc) : __logf(1.f - qc);
        } else if constexpr (MODE == 2){
          float d = v - p.x_ref[(size_t)row * N + col];
          localAcc += d * d;
        } else {
          float zv = bf2f(p.z_src[(size_t)row * N + col]);
          localAcc += zv * (v + p.rbm_h[col]);
        }
      }
      if constexpr (MODE == 0){
        atomicAdd(&sSum[n], ls);
        atomicAdd(&sSq[n],  lsq);
      }
    }
  }
  if constexpr (MODE == 0){
    __syncthreads();
    if (tid < BN && colBlk + tid < N){
      atomicAdd(&p.col_sum[colBlk + tid],   sSum[tid]);
      atomicAdd(&p.col_sumsq[colBlk + tid], sSq[tid]);
    }
  } else {
    atomicAdd(&sRed, localAcc);
    __syncthreads();
    if (tid == 0){
      float* g = (MODE == 1) ? p.logq_accum : (MODE == 2) ? p.recon_accum : p.energy_accum;
      atomicAdd(g, sRed);
    }
  }
}

// ---- 30-step Gibbs chain, rbm_W resident in LDS (bf16), 16 rows per workgroup ----
#define GIBBS_LDS_BYTES ((256 * GLDW + 2 * 16 * GLDW) * 2 + 256 * 4)

__global__ __launch_bounds__(256) void gibbs_kernel(const float* __restrict__ z0,
    const float* __restrict__ rands, const float* __restrict__ rbm_h,
    const u16* __restrict__ Wtr, u16* __restrict__ z_out)
{
  extern __shared__ u16 sm[];
  u16* sW  = sm;                       // [256][GLDW] : W transposed to [n][k]
  u16* zb0 = sW + 256 * GLDW;          // [16][GLDW]
  u16* zb1 = zb0 + 16 * GLDW;
  float* sH = (float*)(zb1 + 16 * GLDW);

  const int tid  = threadIdx.x;
  const int wave = tid >> 5;
  const int lane = tid & 31;
  const int rowBlk = blockIdx.x * 16;

  // stage W (already bf16, [n][k]): thread t copies 16 16B chunks of its n-row, k = 16c..16c+7 (half the row)
  {
    const u16* src = Wtr + (size_t)tid * 256;
    u16* dst = &sW[tid * GLDW];
    #pragma unroll
    for (int c = 0; c < 16; c++) cp16(src + c * 16, dst + c * 16);
  }
  sH[tid] = rbm_h[tid];

  // z_init tile
  const int zr = tid >> 4, zc = (tid & 15) << 4;
  {
    const float* zp = z0 + (size_t)(rowBlk + zr) * 256 + zc;
    #pragma unroll
    for (int c = 0; c < 16; c++) zb0[zr * GLDW + zc + c] = f2bf(zp[c]);
  }
  wait_async();
  __syncthreads();

  u16* cur = zb0; u16* nxt = zb1;
  for (int step = 0; step < NSTEP; step++){
    v16bf af[8];
    const u16* aRow = &cur[(lane & 15) * GLDW];
    #pragma unroll
    for (int kk = 0; kk < 8; kk++) af[kk] = load_a_frag(aRow + kk * 32, lane);

    const float* rp = rands + ((size_t)step * BSZ + rowBlk) * 256;
    #pragma unroll
    for (int tt = 0; tt < 2; tt++){
      const int tile = wave * 2 + tt;
      const int ncol = tile * 16 + (lane & 15);
      const u16* bRow = &sW[ncol * GLDW];
      v8f acc = zero8();
      #pragma unroll
      for (int kk = 0; kk < 8; kk++)
        acc = wmma_bf16(af[kk], load_b_frag(bRow + kk * 32, lane), acc);
      const int rbase = (lane & 16) ? 8 : 0;
      const float hb = sH[ncol];
      #pragma unroll
      for (int i = 0; i < 8; i++){
        int row = rbase + i;
        float t = acc[i] + hb;
        float e = __expf(-t);
        float rr = rp[(size_t)row * 256 + ncol];
        // rr < sigmoid(t)  <=>  rr*(1+e) < 1   (division-free)
        nxt[row * GLDW + ncol] = (rr + rr * e < 1.f) ? (u16)0x3F80 : (u16)0;
      }
    }
    __syncthreads();
    u16* tmp = cur; cur = nxt; nxt = tmp;
  }

  // write final z (bf16) to workspace
  {
    u16* zo = z_out + (size_t)(rowBlk + zr) * 256 + zc;
    *(uint4*)(zo)     = *(const uint4*)(&cur[zr * GLDW + zc]);
    *(uint4*)(zo + 8) = *(const uint4*)(&cur[zr * GLDW + zc + 8]);
  }
}

// one-time: W[K][N] f32 -> Wt[N][K] bf16 (transposed so GEMM B-staging is contiguous)
__global__ void convert_transpose_kernel(const float* __restrict__ W, u16* __restrict__ Wt,
                                         int K, int N){
  int n = blockIdx.x * blockDim.x + threadIdx.x;
  if (n < N){
    for (int k = 0; k < K; k++)
      Wt[(size_t)n * K + k] = f2bf(W[(size_t)k * N + n]);
  }
}

__global__ void init_kernel(float* colSum, float* colSq, float* accums){
  int t = blockIdx.x * blockDim.x + threadIdx.x;
  if (t < 512){ colSum[t] = 0.f; colSq[t] = 0.f; }
  if (t < 8) accums[t] = 0.f;
}

__global__ void bn_finalize_kernel(float* colSum, float* colSq, const float* gamma,
                                   const float* beta, float* scale, float* shift, int H){
  int c = blockIdx.x * blockDim.x + threadIdx.x;
  if (c < H){
    float mu  = colSum[c] * (1.f / (float)BSZ);
    float var = colSq[c] * (1.f / (float)BSZ) - mu * mu;
    float sc  = gamma[c] * rsqrtf(var + 1e-5f);
    scale[c] = sc;
    shift[c] = beta[c] - mu * sc;
    colSum[c] = 0.f; colSq[c] = 0.f;   // ready for decoder pass
  }
}

__global__ __launch_bounds__(256) void bn_relu_kernel(const float* __restrict__ h,
    const float* __restrict__ scale, const float* __restrict__ shift,
    u16* __restrict__ outb, int colMask, size_t total){
  size_t i = (size_t)blockIdx.x * 256 + threadIdx.x;
  if (i < total){
    int col = (int)(i & (size_t)colMask);
    float v = h[i] * scale[col] + shift[col];
    outb[i] = f2bf(v > 0.f ? v : 0.f);
  }
}

__global__ void finalize_kernel(const float* accums, float* out){
  if (threadIdx.x == 0 && blockIdx.x == 0){
    float recon = accums[0] * (1.f / (float)BSZ);
    float logZ  = accums[3] * (1.f / (float)BSZ);
    float kl    = (accums[2] + accums[1]) * (1.f / (float)BSZ) + logZ;
    float elbo  = -recon - 0.001f * kl;
    out[0] = elbo; out[1] = recon; out[2] = kl;
  }
}

extern "C" void kernel_launch(void* const* d_in, const int* in_sizes, int n_in,
                              void* d_out, int out_size, void* d_ws, size_t ws_size,
                              hipStream_t stream)
{
  const float* x       = (const float*)d_in[0];
  const float* rho     = (const float*)d_in[1];
  const float* z_init  = (const float*)d_in[2];
  const float* grands  = (const float*)d_in[3];
  const float* enc_W1  = (const float*)d_in[4];
  const float* enc_b1  = (const float*)d_in[5];
  const float* enc_g1  = (const float*)d_in[6];
  const float* enc_be1 = (const float*)d_in[7];
  const float* enc_W2  = (const float*)d_in[8];
  const float* enc_b2  = (const float*)d_in[9];
  const float* dec_W1  = (const float*)d_in[10];
  const float* dec_b1  = (const float*)d_in[11];
  const float* dec_g1  = (const float*)d_in[12];
  const float* dec_be1 = (const float*)d_in[13];
  const float* dec_W2  = (const float*)d_in[14];
  const float* dec_b2  = (const float*)d_in[15];
  const float* rbm_h   = (const float*)d_in[16];
  const float* rbm_W   = (const float*)d_in[17];

  float* out = (float*)d_out;
  char*  ws  = (char*)d_ws;
  size_t o = 0;
  float* H1    = (float*)(ws + o); o += (size_t)BSZ * HDIM * 4;   // pre-BN activations
  u16*   heb   = (u16*)(ws + o);   o += (size_t)BSZ * HDIM * 2;   // he bf16 (reused as hd)
  u16*   zetab = (u16*)(ws + o);   o += (size_t)BSZ * LDIM * 2;
  u16*   zb    = (u16*)(ws + o);   o += (size_t)BSZ * LDIM * 2;
  u16*   znegb = (u16*)(ws + o);   o += (size_t)BSZ * LDIM * 2;
  u16*   wt_e1 = (u16*)(ws + o);   o += (size_t)HDIM * DIN  * 2;  // enc_W1^T bf16 [512][4000]
  u16*   wt_e2 = (u16*)(ws + o);   o += (size_t)LDIM * HDIM * 2;  // enc_W2^T bf16 [256][512]
  u16*   wt_d1 = (u16*)(ws + o);   o += (size_t)HDIM * LDIM * 2;  // dec_W1^T bf16 [512][256]
  u16*   wt_d2 = (u16*)(ws + o);   o += (size_t)DIN  * HDIM * 2;  // dec_W2^T bf16 [4000][512]
  u16*   wt_r  = (u16*)(ws + o);   o += (size_t)LDIM * LDIM * 2;  // rbm_W^T bf16 [256][256]
  float* colSum  = (float*)(ws + o); o += 512 * 4;
  float* colSq   = (float*)(ws + o); o += 512 * 4;
  float* bnScale = (float*)(ws + o); o += 512 * 4;
  float* bnShift = (float*)(ws + o); o += 512 * 4;
  float* accums  = (float*)(ws + o); o += 64;

  float* z_out    = out + 3;
  float* zeta_out = out + 3 + (size_t)BSZ * LDIM;

  init_kernel<<<2, 256, 0, stream>>>(colSum, colSq, accums);

  // one-time weight convert+transpose to bf16 [N][K]
  convert_transpose_kernel<<<2, 256, 0, stream>>>(enc_W1, wt_e1, DIN,  HDIM);
  convert_transpose_kernel<<<1, 256, 0, stream>>>(enc_W2, wt_e2, HDIM, LDIM);
  convert_transpose_kernel<<<2, 256, 0, stream>>>(dec_W1, wt_d1, LDIM, HDIM);
  convert_transpose_kernel<<<16, 256, 0, stream>>>(dec_W2, wt_d2, HDIM, DIN);
  convert_transpose_kernel<<<1, 256, 0, stream>>>(rbm_W,  wt_r,  LDIM, LDIM);

  EpiParams p{};
  // encoder layer 1: x @ enc_W1 + b1, collect BN stats
  p = {}; p.out_f32 = H1; p.col_sum = colSum; p.col_sumsq = colSq;
  gemm_kernel<false, 0><<<dim3(HDIM / BN, BSZ / BM), 256, 0, stream>>>(
      x, wt_e1, enc_b1, BSZ, HDIM, DIN, p);
  bn_finalize_kernel<<<2, 256, 0, stream>>>(colSum, colSq, enc_g1, enc_be1, bnScale, bnShift, HDIM);
  bn_relu_kernel<<<(BSZ * HDIM) / 256, 256, 0, stream>>>(
      H1, bnScale, bnShift, heb, HDIM - 1, (size_t)BSZ * HDIM);

  // encoder layer 2 + reparameterization (writes z, zeta, entropy)
  p = {}; p.rho = rho; p.z_out = z_out; p.zeta_out = zeta_out;
  p.zeta_bf = zetab; p.z_bf = zb; p.logq_accum = accums + 1;
  gemm_kernel<true, 1><<<dim3(LDIM / BN, BSZ / BM), 256, 0, stream>>>(
      heb, wt_e2, enc_b2, BSZ, LDIM, HDIM, p);

  // decoder layer 1: zeta @ dec_W1 + b1, collect BN stats
  p = {}; p.out_f32 = H1; p.col_sum = colSum; p.col_sumsq = colSq;
  gemm_kernel<true, 0><<<dim3(HDIM / BN, BSZ / BM), 256, 0, stream>>>(
      zetab, wt_d1, dec_b1, BSZ, HDIM, LDIM, p);
  bn_finalize_kernel<<<2, 256, 0, stream>>>(colSum, colSq, dec_g1, dec_be1, bnScale, bnShift, HDIM);
  bn_relu_kernel<<<(BSZ * HDIM) / 256, 256, 0, stream>>>(
      H1, bnScale, bnShift, heb, HDIM - 1, (size_t)BSZ * HDIM);

  // decoder layer 2 fused with recon loss (x_recon never materialized)
  p = {}; p.x_ref = x; p.recon_accum = accums + 0;
  gemm_kernel<true, 2><<<dim3((DIN + BN - 1) / BN, BSZ / BM), 256, 0, stream>>>(
      heb, wt_d2, dec_b2, BSZ, DIN, HDIM, p);

  // positive-phase RBM energy
  p = {}; p.z_src = zb; p.rbm_h = rbm_h; p.energy_accum = accums + 2;
  gemm_kernel<true, 3><<<dim3(LDIM / BN, BSZ / BM), 256, 0, stream>>>(
      zb, wt_r, nullptr, BSZ, LDIM, LDIM, p);

  // 30-step Gibbs chain (W resident in LDS)
  (void)hipFuncSetAttribute(reinterpret_cast<const void*>(gibbs_kernel),
                            hipFuncAttributeMaxDynamicSharedMemorySize, GIBBS_LDS_BYTES);
  gibbs_kernel<<<BSZ / 16, 256, GIBBS_LDS_BYTES, stream>>>(z_init, grands, rbm_h, wt_r, znegb);

  // negative-phase RBM energy
  p = {}; p.z_src = znegb; p.rbm_h = rbm_h; p.energy_accum = accums + 3;
  gemm_kernel<true, 3><<<dim3(LDIM / BN, BSZ / BM), 256, 0, stream>>>(
      znegb, wt_r, nullptr, BSZ, LDIM, LDIM, p);

  finalize_kernel<<<1, 32, 0, stream>>>(accums, out);
}
